// StandardAttention_61753039782615
// MI455X (gfx1250) — matrix operand order repaired and checked by reading, not compiled, not run on hardware
//
#include <hip/hip_runtime.h>

// ---------------- CDNA5 WMMA types / helpers ----------------
typedef __attribute__((ext_vector_type(16))) __bf16 v16bf;
typedef __attribute__((ext_vector_type(8)))  float  v8f;
typedef __attribute__((ext_vector_type(4)))  int    v4i;

union FragU { v16bf v; uint4 u[2]; };

static __device__ __forceinline__ v8f v8zero() {
  v8f z = {0.f,0.f,0.f,0.f,0.f,0.f,0.f,0.f};
  return z;
}

// A-operand fragment (16x32, 16-bit): lane holds row M=lane%16,
// K chunks at hf*8 and 16+hf*8 (hf = lane/16). base = &lds[row*stride + k0]
static __device__ __forceinline__ v16bf ld_frag_a(const __bf16* base, int hf) {
  FragU f;
  f.u[0] = *(const uint4*)(base + hf * 8);
  f.u[1] = *(const uint4*)(base + 16 + hf * 8);
  return f.v;
}
// B-operand fragment (32x16, 16-bit): lane holds column N=lane%16,
// K chunks at hf*8 and 16+hf*8 (hf = lane/16). base = &lds[n*stride + k0]
static __device__ __forceinline__ v16bf ld_frag_b(const __bf16* base, int hf) {
  FragU f;
  f.u[0] = *(const uint4*)(base + hf * 8);
  f.u[1] = *(const uint4*)(base + 16 + hf * 8);
  return f.v;
}

#define WMMA_BF16(a, b, c) \
  __builtin_amdgcn_wmma_f32_16x16x32_bf16(false, (a), false, (b), (short)0, (c), false, false)

// ---- async global->LDS (GLOBAL_LOAD_ASYNC_TO_LDS_B128, ASYNCcnt path) ----
// Builtin params are int4 pointers: (v4i AS1* src, v4i AS3* dst, imm off, imm cpol).
// Generic->AS via integer: shared-aperture generic addresses carry the LDS byte
// offset in addr[31:0]; global generic addresses are the AS1 address.
typedef __attribute__((address_space(1))) v4i g_v4i;
typedef __attribute__((address_space(3))) v4i l_v4i;

static __device__ __forceinline__ void async_ld_b128(const void* g, void* l) {
  __builtin_amdgcn_global_load_async_to_lds_b128(
      (g_v4i*)(uintptr_t)g,
      (l_v4i*)(unsigned int)(uintptr_t)l,
      0, 0);
}
static __device__ __forceinline__ void wait_async0() {
  asm volatile("s_wait_asynccnt 0x0" ::: "memory");
}

// ---------------- problem constants (fixed by harness) ----------------
constexpr int BATCH = 2;
constexpr int SEQ   = 2048;
constexpr int DIM   = 2048;
constexpr int NHEAD = 16;
constexpr int HDIM  = 128;
constexpr int MROWS = BATCH * SEQ;   // 4096

// ---------------- kernel 0a: fp32 -> bf16 convert ----------------
__global__ void cvt_f32_bf16_k(const float* __restrict__ in,
                               __bf16* __restrict__ out, int n) {
  int i = (blockIdx.x * blockDim.x + threadIdx.x) * 8;
  if (i + 8 <= n) {
    const float4* p = (const float4*)(in + i);
    float4 a = p[0], b = p[1];
    union { __bf16 h[8]; uint4 u; } o;
    o.h[0] = (__bf16)a.x; o.h[1] = (__bf16)a.y;
    o.h[2] = (__bf16)a.z; o.h[3] = (__bf16)a.w;
    o.h[4] = (__bf16)b.x; o.h[5] = (__bf16)b.y;
    o.h[6] = (__bf16)b.z; o.h[7] = (__bf16)b.w;
    *(uint4*)(out + i) = o.u;
  }
}

// ---------------- kernel 0b: transpose + convert W (fp32[K][N] -> bf16[N][K]) ----
__global__ void transpose_cvt_k(const float* __restrict__ W,
                                __bf16* __restrict__ Wt) {
  __shared__ float tile[32][33];
  const int tx = threadIdx.x & 31, ty = threadIdx.x >> 5;  // 32x8
  const int c0 = blockIdx.x * 32, r0 = blockIdx.y * 32;
#pragma unroll
  for (int k = 0; k < 4; ++k)
    tile[ty + 8 * k][tx] = W[(size_t)(r0 + ty + 8 * k) * DIM + c0 + tx];
  __syncthreads();
#pragma unroll
  for (int k = 0; k < 4; ++k)
    Wt[(size_t)(c0 + ty + 8 * k) * DIM + r0 + tx] = (__bf16)tile[tx][ty + 8 * k];
}

// ---------------- GEMM: C[m,n] = sum_k A[m,k] * Bt[n,k] -----------------
// 128x128 block tile, 8 waves (4x2), per-wave 32x64 (2x4 WMMA tiles), K-step 64,
// double-buffered LDS fed by async global->LDS loads.
// MODE: 0 = RoPE + bf16 [B,H,S,HD]; 1 = bf16 [B,H,HD,S] (V pre-transposed,
// packed b128 stores); 2 = fp32 row-major.
constexpr int GBT = 128;   // block tile (M and N)
constexpr int GKT = 64;    // K tile
constexpr int GLD = 72;    // padded LDS stride (bf16), 144B: 16B-aligned, conflict-free

template <int MODE>
__global__ __launch_bounds__(256) void gemm_bf16_k(const __bf16* __restrict__ A,
                                                   const __bf16* __restrict__ Bt,
                                                   void* __restrict__ Out) {
  __shared__ __bf16 sA[2][GBT * GLD];
  __shared__ __bf16 sB[2][GBT * GLD];

  const int tid  = threadIdx.x;
  const int lane = tid & 31, w = tid >> 5;
  const int ln = lane & 15, hf = lane >> 4;
  const int wm = w & 3, wn = w >> 2;               // 4x2 wave grid
  const int m0 = blockIdx.y * GBT, n0 = blockIdx.x * GBT;

  // cooperative tile feed: each thread owns 32 bf16 (4 x b128) per array
  const int lrow = tid >> 1, lhalf = tid & 1;
  const __bf16* gA = A  + (size_t)(m0 + lrow) * DIM + lhalf * 32;
  const __bf16* gB = Bt + (size_t)(n0 + lrow) * DIM + lhalf * 32;
  const int lds_off = lrow * GLD + lhalf * 32;

  auto feed = [&](int buf, int kb) {
#pragma unroll
    for (int i = 0; i < 4; ++i) {
      async_ld_b128(gA + kb * GKT + i * 8, &sA[buf][lds_off + i * 8]);
      async_ld_b128(gB + kb * GKT + i * 8, &sB[buf][lds_off + i * 8]);
    }
  };

  feed(0, 0);
  wait_async0();
  __syncthreads();

  v8f acc[2][4];
#pragma unroll
  for (int mi = 0; mi < 2; ++mi)
#pragma unroll
    for (int ni = 0; ni < 4; ++ni) acc[mi][ni] = v8zero();

  const int NKB = DIM / GKT;  // 32
  for (int kb = 0; kb < NKB; ++kb) {
    const int cur = kb & 1;
    if (kb + 1 < NKB) feed(cur ^ 1, kb + 1);
#pragma unroll
    for (int kk = 0; kk < 2; ++kk) {
      v16bf af[2], bfm[4];
#pragma unroll
      for (int mi = 0; mi < 2; ++mi)
        af[mi] = ld_frag_a(&sA[cur][(wm * 32 + mi * 16 + ln) * GLD + kk * 32], hf);
#pragma unroll
      for (int ni = 0; ni < 4; ++ni)
        bfm[ni] = ld_frag_b(&sB[cur][(wn * 64 + ni * 16 + ln) * GLD + kk * 32], hf);
#pragma unroll
      for (int mi = 0; mi < 2; ++mi)
#pragma unroll
        for (int ni = 0; ni < 4; ++ni)
          acc[mi][ni] = WMMA_BF16(af[mi], bfm[ni], acc[mi][ni]);
    }
    if (kb + 1 < NKB) {
      wait_async0();
      __syncthreads();
    }
  }

  // epilogue: C/D layout -- lane holds M = r + 8*hf, N = ln (per v8f element r)
#pragma unroll
  for (int mi = 0; mi < 2; ++mi)
#pragma unroll
    for (int ni = 0; ni < 4; ++ni) {
      const int gm0 = m0 + wm * 32 + mi * 16 + 8 * hf;   // row for r=0
      const int gn  = n0 + wn * 64 + ni * 16 + ln;
      if (MODE == 2) {
#pragma unroll
        for (int r = 0; r < 8; ++r)
          ((float*)Out)[(size_t)(gm0 + r) * DIM + gn] = acc[mi][ni][r];
      } else if (MODE == 1) {
        // V: store pre-transposed [B,H,HD,S]; 8 consecutive s -> one b128
        union { __bf16 h[8]; uint4 u; } o;
#pragma unroll
        for (int r = 0; r < 8; ++r) o.h[r] = (__bf16)acc[mi][ni][r];
        const int b = gm0 >> 11, s0 = gm0 & (SEQ - 1);
        const int hh = gn >> 7, hd = gn & (HDIM - 1);
        *(uint4*)&((__bf16*)Out)[(((size_t)(b * NHEAD + hh)) * HDIM + hd) * SEQ + s0] = o.u;
      } else {
        // Q/K: fused RoPE, store [B,H,S,HD]
        const int d2 = (gn & (HDIM - 1)) >> 1;
        const float freq = __expf((float)d2 * -0.14391156857277346f); // 10000^(-2*d2/128)
        const int hh = gn >> 7, hd = gn & (HDIM - 1);
#pragma unroll
        for (int r = 0; r < 8; ++r) {
          const int gm = gm0 + r;
          float v = acc[mi][ni][r];
          float pv = __shfl_xor(v, 1);  // even/odd pair lives in lanes (ln, ln^1)
          const float ang = (float)(gm & (SEQ - 1)) * freq;
          const float cs = __cosf(ang), sn = __sinf(ang);
          v = (gn & 1) ? (v * cs + pv * sn) : (v * cs - pv * sn);
          const int s = gm & (SEQ - 1);
          ((__bf16*)Out)[(((size_t)((gm >> 11) * NHEAD + hh)) * SEQ + s) * HDIM + hd] = (__bf16)v;
        }
      }
    }
}

// ---------------- flash attention (causal, online softmax) ----------------
// grid: (S/128 q-blocks, B*H). 8 waves; wave w owns q-rows [w*16, w*16+16).
// K tiles [64][128] row-major (== WMMA Bt layout); V arrives pre-transposed
// from global [B,H,HD,S] so its tile [128][64] is loaded straight into the
// B-operand layout. Both double-buffered via async global->LDS.
constexpr int FLQ = 136;  // padded stride for Q/K tiles (128+8)
constexpr int FLV = 72;   // padded stride for Vt / P tiles (64+8)

__global__ __launch_bounds__(256) void flash_attn_k(const __bf16* __restrict__ Qg,
                                                    const __bf16* __restrict__ Kg,
                                                    const __bf16* __restrict__ Vtg,
                                                    __bf16* __restrict__ Oa) {
  __shared__ __bf16 sQ[128 * FLQ];          // Q block 128 x 128
  __shared__ __bf16 sK[2][64 * FLQ];        // K tile 64 x 128
  __shared__ __bf16 sVt[2][128 * FLV];      // V tile transposed [d][kpos]
  __shared__ __bf16 sP[8][16 * FLV];        // per-wave P scratch 16 x 64

  const int tid = threadIdx.x, lane = tid & 31, w = tid >> 5;
  const int ln = lane & 15, hf = lane >> 4;
  const int qb = blockIdx.x, bh = blockIdx.y;
  const size_t base  = (size_t)bh * SEQ * HDIM;   // Q/K: [bh][s][d]
  const size_t baseV = (size_t)bh * HDIM * SEQ;   // Vt:  [bh][d][s]

  {  // Q block: async, 8 x b128 per thread
    const int row = tid >> 1, seg = tid & 1;
    const __bf16* p = Qg + base + (size_t)(qb * 128 + row) * HDIM + seg * 64;
    __bf16* l = &sQ[row * FLQ + seg * 64];
#pragma unroll
    for (int i = 0; i < 8; ++i) async_ld_b128(p + i * 8, l + i * 8);
  }

  // KV tile feed (async): K row-major rows, Vt d-major rows
  const int krow = tid >> 2, kseg = tid & 3;        // K: 64 rows x 4 segs of 32
  const int vrow = tid >> 1, vseg = tid & 1;        // Vt: 128 rows x 2 segs of 32
  auto feedKV = [&](int buf, int j) {
    const __bf16* pk = Kg + base + (size_t)(j * 64 + krow) * HDIM + kseg * 32;
    __bf16* lk = &sK[buf][krow * FLQ + kseg * 32];
    const __bf16* pv = Vtg + baseV + (size_t)vrow * SEQ + j * 64 + vseg * 32;
    __bf16* lv = &sVt[buf][vrow * FLV + vseg * 32];
#pragma unroll
    for (int i = 0; i < 4; ++i) {
      async_ld_b128(pk + i * 8, lk + i * 8);
      async_ld_b128(pv + i * 8, lv + i * 8);
    }
  };

  feedKV(0, 0);
  wait_async0();
  __syncthreads();

  float m_run[8], l_run[8];
  v8f oacc[8];
#pragma unroll
  for (int r = 0; r < 8; ++r) { m_run[r] = -1e30f; l_run[r] = 0.f; }
#pragma unroll
  for (int t = 0; t < 8; ++t) oacc[t] = v8zero();

  const float scale = 0.08838834764831845f;  // 1/sqrt(128)
  const int nj = 2 * qb + 2;                 // causal KV range
  for (int j = 0; j < nj; ++j) {
    const int cur = j & 1;
    if (j + 1 < nj) feedKV(cur ^ 1, j + 1);

    // scores: 16 (q rows of this wave) x 64 (kpos) = 4 WMMA tiles, K over d
    v8f sc[4];
#pragma unroll
    for (int t = 0; t < 4; ++t) sc[t] = v8zero();
#pragma unroll
    for (int dk = 0; dk < 4; ++dk) {
      v16bf aq = ld_frag_a(&sQ[(w * 16 + ln) * FLQ + dk * 32], hf);
      v16bf bk[4];
#pragma unroll
      for (int t = 0; t < 4; ++t)
        bk[t] = ld_frag_b(&sK[cur][(t * 16 + ln) * FLQ + dk * 32], hf);
#pragma unroll
      for (int t = 0; t < 4; ++t) sc[t] = WMMA_BF16(aq, bk[t], sc[t]);
    }

    // scale + causal mask
#pragma unroll
    for (int t = 0; t < 4; ++t)
#pragma unroll
      for (int r = 0; r < 8; ++r) {
        const int kp = j * 64 + t * 16 + ln;
        const int qr = qb * 128 + w * 16 + r + 8 * hf;
        const float v = sc[t][r] * scale;
        sc[t][r] = (kp > qr) ? -1e30f : v;
      }

    // online softmax (rows replicated over 16-lane halves; reduce over ln)
    float al[8];
#pragma unroll
    for (int r = 0; r < 8; ++r) {
      float mx = fmaxf(fmaxf(sc[0][r], sc[1][r]), fmaxf(sc[2][r], sc[3][r]));
#pragma unroll
      for (int d = 1; d < 16; d <<= 1) mx = fmaxf(mx, __shfl_xor(mx, d));
      const float nm = fmaxf(m_run[r], mx);
      al[r] = __expf(m_run[r] - nm);
      m_run[r] = nm;
      float rs = 0.f;
#pragma unroll
      for (int t = 0; t < 4; ++t) { float p = __expf(sc[t][r] - nm); sc[t][r] = p; rs += p; }
#pragma unroll
      for (int d = 1; d < 16; d <<= 1) rs += __shfl_xor(rs, d);
      l_run[r] = l_run[r] * al[r] + rs;
    }
#pragma unroll
    for (int t = 0; t < 8; ++t)
#pragma unroll
      for (int r = 0; r < 8; ++r) oacc[t][r] *= al[r];

    // P: C-layout -> LDS -> A-layout (per-wave region, LDS pipe is in-order)
#pragma unroll
    for (int t = 0; t < 4; ++t)
#pragma unroll
      for (int r = 0; r < 8; ++r)
        sP[w][(r + 8 * hf) * FLV + t * 16 + ln] = (__bf16)sc[t][r];

    // O += P(16x64) @ V(64x128)
#pragma unroll
    for (int t2 = 0; t2 < 2; ++t2) {
      v16bf ap = ld_frag_a(&sP[w][ln * FLV + t2 * 32], hf);
      v16bf bv[4];
#pragma unroll
      for (int g = 0; g < 2; ++g) {
#pragma unroll
        for (int td = 0; td < 4; ++td)
          bv[td] = ld_frag_b(&sVt[cur][((g * 4 + td) * 16 + ln) * FLV + t2 * 32], hf);
#pragma unroll
        for (int td = 0; td < 4; ++td)
          oacc[g * 4 + td] = WMMA_BF16(ap, bv[td], oacc[g * 4 + td]);
      }
    }

    if (j + 1 < nj) {
      wait_async0();
      __syncthreads();
    }
  }

  // normalize + store attn output as bf16 row-major [B*S, DIM]
  const int b = bh >> 4, h = bh & 15;
#pragma unroll
  for (int r = 0; r < 8; ++r) {
    const float rl = 1.0f / l_run[r];
    const int gs = qb * 128 + w * 16 + r + 8 * hf;
#pragma unroll
    for (int td = 0; td < 8; ++td) {
      const int gd = h * HDIM + td * 16 + ln;
      Oa[((size_t)(b * SEQ) + gs) * DIM + gd] = (__bf16)(oacc[td][r] * rl);
    }
  }
}

// ---------------- host ----------------
extern "C" void kernel_launch(void* const* d_in, const int* in_sizes, int n_in,
                              void* d_out, int out_size, void* d_ws, size_t ws_size,
                              hipStream_t stream) {
  const float* x  = (const float*)d_in[0];
  const float* Wq = (const float*)d_in[1];
  const float* Wk = (const float*)d_in[2];
  const float* Wv = (const float*)d_in[3];
  const float* Wo = (const float*)d_in[4];
  float* out = (float*)d_out;

  char* ws = (char*)d_ws;
  __bf16* xb  = (__bf16*)(ws);                             // 16 MB (reused as attn)
  __bf16* wqT = (__bf16*)(ws + (size_t)16 * 1024 * 1024);  // 8 MB each
  __bf16* wkT = (__bf16*)(ws + (size_t)24 * 1024 * 1024);
  __bf16* wvT = (__bf16*)(ws + (size_t)32 * 1024 * 1024);
  __bf16* woT = (__bf16*)(ws + (size_t)40 * 1024 * 1024);
  __bf16* Qr  = (__bf16*)(ws + (size_t)48 * 1024 * 1024);  // 16 MB each
  __bf16* Kr  = (__bf16*)(ws + (size_t)64 * 1024 * 1024);
  __bf16* Vtr = (__bf16*)(ws + (size_t)80 * 1024 * 1024);  // [B,H,HD,S]
  __bf16* Att = xb;  // safe reuse: xb fully consumed before flash writes

  cvt_f32_bf16_k<<<(MROWS * DIM) / (256 * 8), 256, 0, stream>>>(x, xb, MROWS * DIM);
  dim3 tg(DIM / 32, DIM / 32);
  transpose_cvt_k<<<tg, 256, 0, stream>>>(Wq, wqT);
  transpose_cvt_k<<<tg, 256, 0, stream>>>(Wk, wkT);
  transpose_cvt_k<<<tg, 256, 0, stream>>>(Wv, wvT);
  transpose_cvt_k<<<tg, 256, 0, stream>>>(Wo, woT);

  dim3 gg(DIM / GBT, MROWS / GBT);  // (16, 32)
  gemm_bf16_k<0><<<gg, 256, 0, stream>>>(xb, wqT, (void*)Qr);
  gemm_bf16_k<0><<<gg, 256, 0, stream>>>(xb, wkT, (void*)Kr);
  gemm_bf16_k<1><<<gg, 256, 0, stream>>>(xb, wvT, (void*)Vtr);

  dim3 fg(SEQ / 128, BATCH * NHEAD);  // (16, 32)
  flash_attn_k<<<fg, 256, 0, stream>>>(Qr, Kr, Vtr, Att);

  gemm_bf16_k<2><<<gg, 256, 0, stream>>>(Att, woT, (void*)out);
}
